// ReverseEmbedding_730144440366
// MI455X (gfx1250) — compile-verified, matrix operand order checked by reading, not run on hardware
//
#include <hip/hip_runtime.h>

// ---------------------------------------------------------------------------
// ReverseEmbedding: argmax_v cos(x_n, W_v) -> one-hot [B,S,V]
//   x: [2,4096,64] f32   W: [32000,64] f32   out: [8192, 32000] f32 one-hot
//
// MI455X plan (23.3 TB/s HBM, 192 MB L2, wave32 WMMA):
//  * HBM floor = 1.05 GB one-hot stores (~45us). Fuse the zero-fill into the
//    argmax kernel (each WG zeros its own 32-token slice, 4 x 128-bit NT
//    stores per thread per vocab iteration) so it overlaps WMMA compute
//    instead of being a separate serialized pass.
//  * |x| is argmax-invariant; |W_v| is not -> pre-normalize W rows once and
//    split into bf16 hi/lo planes (bf16x3 trick: hh + hl + lh => ~2^-16 rel
//    error, argmax-faithful to f32) stored in d_ws (8 MB, L2-resident).
//  * 32 tokens per WG (2 M-tiles per wave, shared B regs) halves L2 read
//    amplification vs 16-token tiles: 256 WGs x 8.2 MB = 2.1 GB from L2.
//  * 12 v_wmma_f32_16x16x32_bf16 per 16-row vocab chunk; register argmax,
//    shfl_xor + LDS reduction, s_wait_storecnt + barrier, scatter 1.0f.
// ---------------------------------------------------------------------------

typedef __attribute__((ext_vector_type(16))) __bf16 v16bf;
typedef __attribute__((ext_vector_type(8)))  float  v8f;
typedef __attribute__((ext_vector_type(4)))  float  v4f;   // clang vector: OK for NT stores

#define VOCAB   32000
#define EMB     64
#define NTOK    8192
#define TOK_WG  32                     // tokens per workgroup (2 tiles of 16)
#define NCHUNK  (VOCAB / (16 * 8))     // 250 vocab chunks per wave

static __device__ __forceinline__ unsigned short bf_bits(__bf16 h) {
  return __builtin_bit_cast(unsigned short, h);
}

// One wave per vocab row: normalize, split into bf16 hi/lo planes in ws.
__global__ void __launch_bounds__(256) prep_w_kernel(
    const float* __restrict__ W,
    unsigned int* __restrict__ Wh,   // hi plane, 2 bf16 per uint, [VOCAB*32]
    unsigned int* __restrict__ Wl) { // lo plane
  const int lane = threadIdx.x & 31;
  const int wave = threadIdx.x >> 5;
  const int v    = blockIdx.x * 8 + wave;   // grid = VOCAB/8 exactly

  float2 w2 = *(const float2*)(W + (size_t)v * EMB + lane * 2);
  float ss = w2.x * w2.x + w2.y * w2.y;
  ss += __shfl_xor(ss, 1);
  ss += __shfl_xor(ss, 2);
  ss += __shfl_xor(ss, 4);
  ss += __shfl_xor(ss, 8);
  ss += __shfl_xor(ss, 16);
  const float rn = rsqrtf(ss);

  const float f0 = w2.x * rn, f1 = w2.y * rn;
  const __bf16 h0 = (__bf16)f0, h1 = (__bf16)f1;
  const __bf16 l0 = (__bf16)(f0 - (float)h0);
  const __bf16 l1 = (__bf16)(f1 - (float)h1);

  const size_t o = (size_t)v * 32 + lane;
  Wh[o] = ((unsigned int)bf_bits(h1) << 16) | bf_bits(h0);
  Wl[o] = ((unsigned int)bf_bits(l1) << 16) | bf_bits(l0);
}

// One workgroup (8 wave32) per 32-token tile; scans all 32000 vocab rows,
// zero-fills its own output slice along the way, then scatters the ones.
__global__ void __launch_bounds__(256) argmax_onehot_kernel(
    const float* __restrict__ X,
    const unsigned short* __restrict__ Wh,
    const unsigned short* __restrict__ Wl,
    float* __restrict__ out) {
  __shared__ float sV[2][8][16];
  __shared__ int   sI[2][8][16];

  const int lane = threadIdx.x & 31;
  const int wave = threadIdx.x >> 5;
  const int half = lane >> 4;    // K-half selector per ISA 16-bit layouts
  const int mn   = lane & 15;    // row (A) / col (B) index within tile
  const int t0   = blockIdx.x * TOK_WG;

  // A tiles (2 x 16 tokens x 64 dims) in the 16-bit A-matrix VGPR layout,
  // hi/lo split. Element j of chunk c holds K = j + (j>=8?8:0) + half*8 + c*32.
  v16bf Ah[2][2], Al[2][2];
#pragma unroll
  for (int tl = 0; tl < 2; ++tl) {
#pragma unroll
    for (int c = 0; c < 2; ++c) {
#pragma unroll
      for (int j = 0; j < 16; ++j) {
        const int k = j + ((j >= 8) ? 8 : 0) + half * 8 + c * 32;
        const float xv = X[(size_t)(t0 + tl * 16 + mn) * EMB + k];
        const __bf16 h = (__bf16)xv;
        Ah[tl][c][j] = h;
        Al[tl][c][j] = (__bf16)(xv - (float)h);
      }
    }
  }

  float bestV[2][8];
  int   bestI[2][8];
#pragma unroll
  for (int tl = 0; tl < 2; ++tl)
#pragma unroll
    for (int r = 0; r < 8; ++r) { bestV[tl][r] = -3.0e38f; bestI[tl][r] = 0; }

  // Zero-fill region owned by this WG: 32*32000 floats = 256000 v4f,
  // emitted as exactly 4 NT stores per thread per vocab iteration.
  v4f* __restrict__ out4 = (v4f*)(out + (size_t)t0 * VOCAB);
  const v4f z4 = {0.f, 0.f, 0.f, 0.f};

  for (int it = 0; it < NCHUNK; ++it) {
#pragma unroll
    for (int s = 0; s < 4; ++s)
      __builtin_nontemporal_store(z4, &out4[(size_t)it * 1024 + s * 256 + threadIdx.x]);

    const int vbase = wave * 16 + it * 128;
    const int n = vbase + mn;  // this lane's vocab row within the chunk
    const uint4* bh = (const uint4*)(Wh + (size_t)n * EMB);  // row = 8 x uint4
    const uint4* bl = (const uint4*)(Wl + (size_t)n * EMB);

    v8f acc0 = {0.f, 0.f, 0.f, 0.f, 0.f, 0.f, 0.f, 0.f};
    v8f acc1 = {0.f, 0.f, 0.f, 0.f, 0.f, 0.f, 0.f, 0.f};
#pragma unroll
    for (int c = 0; c < 2; ++c) {
      // B lane layout mirrors A: elements 0..7 -> K span [half*8 + c*32, +8),
      // elements 8..15 -> K span [16 + half*8 + c*32, +8). 8 bf16 = one uint4.
      const int q0 = (half * 8 + c * 32) >> 3;
      const int q1 = (16 + half * 8 + c * 32) >> 3;
      v16bf Bh, Bl;
      ((uint4*)&Bh)[0] = bh[q0];
      ((uint4*)&Bh)[1] = bh[q1];
      ((uint4*)&Bl)[0] = bl[q0];
      ((uint4*)&Bl)[1] = bl[q1];

      acc0 = __builtin_amdgcn_wmma_f32_16x16x32_bf16(false, Ah[0][c], false, Bh,
                                                     (short)0, acc0, false, false);
      acc1 = __builtin_amdgcn_wmma_f32_16x16x32_bf16(false, Ah[1][c], false, Bh,
                                                     (short)0, acc1, false, false);
      acc0 = __builtin_amdgcn_wmma_f32_16x16x32_bf16(false, Ah[0][c], false, Bl,
                                                     (short)0, acc0, false, false);
      acc1 = __builtin_amdgcn_wmma_f32_16x16x32_bf16(false, Ah[1][c], false, Bl,
                                                     (short)0, acc1, false, false);
      acc0 = __builtin_amdgcn_wmma_f32_16x16x32_bf16(false, Al[0][c], false, Bh,
                                                     (short)0, acc0, false, false);
      acc1 = __builtin_amdgcn_wmma_f32_16x16x32_bf16(false, Al[1][c], false, Bh,
                                                     (short)0, acc1, false, false);
    }

    // C layout: VGPR r, lane L -> score[M = r + half*8][N = lane&15].
    // Within a lane, vocab index strictly increases -> '>' keeps first max.
#pragma unroll
    for (int r = 0; r < 8; ++r) {
      const float s0 = acc0[r];
      if (s0 > bestV[0][r]) { bestV[0][r] = s0; bestI[0][r] = n; }
      const float s1 = acc1[r];
      if (s1 > bestV[1][r]) { bestV[1][r] = s1; bestI[1][r] = n; }
    }
  }

  // Reduce across the 16 lanes of each half (ties -> lower vocab index).
#pragma unroll
  for (int tl = 0; tl < 2; ++tl) {
#pragma unroll
    for (int r = 0; r < 8; ++r) {
#pragma unroll
      for (int off = 1; off < 16; off <<= 1) {
        const float oV = __shfl_xor(bestV[tl][r], off);
        const int   oI = __shfl_xor(bestI[tl][r], off);
        if (oV > bestV[tl][r] || (oV == bestV[tl][r] && oI < bestI[tl][r])) {
          bestV[tl][r] = oV; bestI[tl][r] = oI;
        }
      }
    }
  }
  if (lane == 0 || lane == 16) {
    const int mb = half * 8;
#pragma unroll
    for (int tl = 0; tl < 2; ++tl)
#pragma unroll
      for (int r = 0; r < 8; ++r) {
        sV[tl][wave][mb + r] = bestV[tl][r];
        sI[tl][wave][mb + r] = bestI[tl][r];
      }
  }

  // All zero stores must land before the scatter (cross-wave ordering):
  // drain this wave's stores, then workgroup barrier.
  asm volatile("s_wait_storecnt 0x0" ::: "memory");
  __syncthreads();

  // Final cross-wave reduce + scatter of the 1.0f into the zeroed slice.
  if (threadIdx.x < 32) {
    const int m  = threadIdx.x & 15;
    const int tl = threadIdx.x >> 4;
    float bv = sV[tl][0][m];
    int   bi = sI[tl][0][m];
#pragma unroll
    for (int w = 1; w < 8; ++w) {
      const float v = sV[tl][w][m];
      const int   i = sI[tl][w][m];
      if (v > bv || (v == bv && i < bi)) { bv = v; bi = i; }
    }
    __builtin_nontemporal_store(1.0f, &out[(size_t)(t0 + tl * 16 + m) * VOCAB + bi]);
  }
}

extern "C" void kernel_launch(void* const* d_in, const int* in_sizes, int n_in,
                              void* d_out, int out_size, void* d_ws, size_t ws_size,
                              hipStream_t stream) {
  (void)in_sizes; (void)n_in; (void)ws_size; (void)out_size;
  const float* X = (const float*)d_in[0];
  const float* W = (const float*)d_in[1];
  float* out = (float*)d_out;

  // Workspace layout: bf16 hi plane then lo plane (raw bits), 4.0 MB each.
  unsigned short* Wh = (unsigned short*)d_ws;
  unsigned short* Wl = Wh + (size_t)VOCAB * EMB;

  // 1) normalize + bf16 hi/lo split of W (L2-resident, 8 MB)
  prep_w_kernel<<<VOCAB / 8, 256, 0, stream>>>(W, (unsigned int*)Wh, (unsigned int*)Wl);
  // 2) fused zero-fill + WMMA argmax + scatter ones
  argmax_onehot_kernel<<<NTOK / TOK_WG, 256, 0, stream>>>(X, Wh, Wl, out);
}
